// DisentangledCodebooks_56049323213428
// MI455X (gfx1250) — matrix operand order repaired and checked by reading, not compiled, not run on hardware
//
#include <hip/hip_runtime.h>

// ---------------------------------------------------------------------------
// Types for CDNA5 WMMA (wave32): bf16 A/B fragments (16 x bf16 per lane),
// f32 accumulator (8 x f32 per lane).
// ---------------------------------------------------------------------------
typedef __attribute__((ext_vector_type(16))) __bf16 v16bf;
typedef __attribute__((ext_vector_type(8)))  __bf16 v8bf;
typedef __attribute__((ext_vector_type(8)))  float  v8f;
typedef __attribute__((ext_vector_type(4)))  int    v4i;

union V16U { v16bf v; v8bf h[2]; };
__device__ __forceinline__ v16bf make16(v8bf lo, v8bf hi) {
  V16U u; u.h[0] = lo; u.h[1] = hi; return u.v;
}

#define BM 64
#define BN 64
#define BK 32
#define LDT 40   // LDS row stride in bf16 elements (multiple of 8 -> 16B aligned rows)

// ---------------------------------------------------------------------------
// Async global->LDS (CDNA5 GLOBAL_LOAD_ASYNC_TO_LDS_B128, ASYNCcnt-tracked).
// Builtin signature (from clang diagnostic): (int4 AS1*, int4 AS3*, Ii, Ii).
// ---------------------------------------------------------------------------
#if __has_builtin(__builtin_amdgcn_global_load_async_to_lds_b128)
#define HAVE_ASYNC_LDS 1
#endif

#if defined(HAVE_ASYNC_LDS)
typedef __attribute__((address_space(1))) v4i* gptr_v4i;
typedef __attribute__((address_space(3))) v4i* lptr_v4i;
__device__ __forceinline__ void async_cp16(const __bf16* g, __bf16* l) {
  __builtin_amdgcn_global_load_async_to_lds_b128(
      (gptr_v4i)(g), (lptr_v4i)(l), 0, 0);
}
#if __has_builtin(__builtin_amdgcn_s_wait_asynccnt)
__device__ __forceinline__ void wait_async0() { __builtin_amdgcn_s_wait_asynccnt(0); }
__device__ __forceinline__ void wait_async4() { __builtin_amdgcn_s_wait_asynccnt(4); }
#else
__device__ __forceinline__ void wait_async0() { asm volatile("s_wait_asynccnt 0x0" ::: "memory"); }
__device__ __forceinline__ void wait_async4() { asm volatile("s_wait_asynccnt 0x4" ::: "memory"); }
#endif
#else
__device__ __forceinline__ void wait_async0() {}
__device__ __forceinline__ void wait_async4() {}
#endif

// Issue the 4 x 16B copies (A row seg x2, B row seg x2) for one K-tile into
// LDS buffer p. With async hardware this is 4 ASYNC ops and no VGPR staging;
// the fallback stages all four chunks in registers first (4 loads -> 1 wait ->
// 4 stores instead of 4 serialized load/store pairs).
__device__ __forceinline__ void issue_tile(const __bf16* ag, const __bf16* bg,
                                           __bf16* la, __bf16* lb) {
#if defined(HAVE_ASYNC_LDS)
  async_cp16(ag,     la);
  async_cp16(ag + 8, la + 8);
  async_cp16(bg,     lb);
  async_cp16(bg + 8, lb + 8);
#else
  v8bf a0 = *(const v8bf*)(ag);
  v8bf a1 = *(const v8bf*)(ag + 8);
  v8bf b0 = *(const v8bf*)(bg);
  v8bf b1 = *(const v8bf*)(bg + 8);
  *(v8bf*)(la)     = a0;
  *(v8bf*)(la + 8) = a1;
  *(v8bf*)(lb)     = b0;
  *(v8bf*)(lb + 8) = b1;
#endif
}

// ---------------------------------------------------------------------------
// Tiled bf16 WMMA GEMM:  C(MxN) = A(MxK) * B(KxN) (+bias) (optional ReLU)
// A row-major (lda), B supplied PRE-TRANSPOSED row-major NxK (ldb = K stride).
// Optionally writes f32 out (ldc) and/or bf16 out (ldcb).
// Block = 128 threads (4 waves); block tile 64x64; each wave owns 32x32
// as a 2x2 grid of v_wmma_f32_16x16x32_bf16 tiles. Double-buffered LDS fed
// by async global->LDS copies: tile t+1 is in flight during WMMA on tile t.
// ---------------------------------------------------------------------------
__global__ __launch_bounds__(128)
void gemm_bf16_wmma(const __bf16* __restrict__ A, int lda,
                    const __bf16* __restrict__ Bt, int ldb,
                    const float* __restrict__ bias,
                    float* __restrict__ Cf, int ldc,
                    __bf16* __restrict__ Cb, int ldcb,
                    int M, int N, int K, int relu)
{
  __shared__ __bf16 As[2][BM][LDT];
  __shared__ __bf16 Bs[2][BN][LDT];

  const int tid   = threadIdx.x;
  const int lane  = tid & 31;
  const int wid   = tid >> 5;
  const int hf    = lane >> 4;      // half-wave: K-subgroup in frags, row-subgroup in acc
  const int l16   = lane & 15;
  const int bm    = blockIdx.y * BM;
  const int bn    = blockIdx.x * BN;
  const int waveM = (wid >> 1) * 32;
  const int waveN = (wid & 1) * 32;

  const int lrow = tid >> 1;          // 0..63 : tile row loaded by this thread
  const int lseg = (tid & 1) * 16;    // 0 or 16 : k-segment

  const __bf16* aRow = A  + (size_t)(bm + lrow) * lda + lseg;
  const __bf16* bRow = Bt + (size_t)(bn + lrow) * ldb + lseg;

  v8f acc[2][2] = {};

  const int T = K / BK;
  issue_tile(aRow, bRow, &As[0][lrow][lseg], &Bs[0][lrow][lseg]);

  for (int t = 0; t < T; ++t) {
    const int p = t & 1;
    if (t + 1 < T) {
      const int k1 = (t + 1) * BK;
      issue_tile(aRow + k1, bRow + k1, &As[p ^ 1][lrow][lseg], &Bs[p ^ 1][lrow][lseg]);
      wait_async4();                 // our tile-t copies have landed
    } else {
      wait_async0();
    }
    __syncthreads();                 // all waves' tile-t data visible

    v16bf afrag[2], bfrag[2];
#pragma unroll
    for (int mi = 0; mi < 2; ++mi) {
      const __bf16* ap = &As[p][waveM + mi * 16 + l16][0];
      // A 16x32 bf16 layout: lane half selects K in {8h..} and {16+8h..}
      afrag[mi] = make16(*(const v8bf*)(ap + 8 * hf),
                         *(const v8bf*)(ap + 16 + 8 * hf));
    }
#pragma unroll
    for (int ni = 0; ni < 2; ++ni) {
      const __bf16* bp = &Bs[p][waveN + ni * 16 + l16][0];
      // B 32x16 bf16 layout: lane half selects K in {8h..} and {16+8h..}
      bfrag[ni] = make16(*(const v8bf*)(bp + 8 * hf),
                         *(const v8bf*)(bp + 16 + 8 * hf));
    }
#pragma unroll
    for (int mi = 0; mi < 2; ++mi)
#pragma unroll
      for (int ni = 0; ni < 2; ++ni)
        acc[mi][ni] = __builtin_amdgcn_wmma_f32_16x16x32_bf16(
            false, afrag[mi], false, bfrag[ni], (short)0, acc[mi][ni],
            false, false);
    __syncthreads();                 // buffer p free for tile t+2's copies
  }

  // Epilogue: C-layout lane(l16)=col, VGPR v + 8*half = row within 16x16 tile.
#pragma unroll
  for (int mi = 0; mi < 2; ++mi)
#pragma unroll
    for (int ni = 0; ni < 2; ++ni) {
      const int col = bn + waveN + ni * 16 + l16;
      const float bv = bias[col];
#pragma unroll
      for (int v = 0; v < 8; ++v) {
        const int row = bm + waveM + mi * 16 + hf * 8 + v;
        float x = acc[mi][ni][v] + bv;
        if (relu) x = fmaxf(x, 0.0f);
        if (Cf) Cf[(size_t)row * ldc + col] = x;
        if (Cb) Cb[(size_t)row * ldcb + col] = (__bf16)x;
      }
    }
}

// ---------------------------------------------------------------------------
// Fused VQ argmin:  scores S = z @ cb^T via WMMA, argmin_k( ||c_k||^2 - 2 S )
// (||z||^2 dropped: constant per row). Codebook padded to Kpad (mult of 64);
// pad rows are zero with norm +3e38 so they never win.
// Block handles 64 z-rows; loops over all codebook 64-column tiles keeping
// running min/argmin in registers; cross-lane argmin via shfl_xor.
// Same async double-buffered LDS pipeline as the GEMM.
// ---------------------------------------------------------------------------
__global__ __launch_bounds__(128)
void vq_argmin_wmma(const __bf16* __restrict__ Z,    // rows x 256
                    const __bf16* __restrict__ CB,   // Kpad x 256 (bf16)
                    const float* __restrict__ cbn,   // Kpad norms
                    int* __restrict__ idxOut,
                    int Kpad)
{
  __shared__ __bf16 As[2][BM][LDT];
  __shared__ __bf16 Bs[2][BN][LDT];

  const int tid   = threadIdx.x;
  const int lane  = tid & 31;
  const int wid   = tid >> 5;
  const int hf    = lane >> 4;
  const int l16   = lane & 15;
  const int bm    = blockIdx.x * BM;
  const int waveM = (wid >> 1) * 32;
  const int waveN = (wid & 1) * 32;
  const int lrow  = tid >> 1;
  const int lseg  = (tid & 1) * 16;

  const __bf16* aRow = Z + (size_t)(bm + lrow) * 256 + lseg;

  float best[2][8];
  int   bidx[2][8];
#pragma unroll
  for (int mi = 0; mi < 2; ++mi)
#pragma unroll
    for (int v = 0; v < 8; ++v) { best[mi][v] = 3.4e38f; bidx[mi][v] = 0; }

  for (int n0 = 0; n0 < Kpad; n0 += BN) {
    const __bf16* bRow = CB + (size_t)(n0 + lrow) * 256 + lseg;
    v8f acc[2][2] = {};

    issue_tile(aRow, bRow, &As[0][lrow][lseg], &Bs[0][lrow][lseg]);
    for (int t = 0; t < 8; ++t) {            // 256 / BK
      const int p = t & 1;
      if (t + 1 < 8) {
        const int k1 = (t + 1) * BK;
        issue_tile(aRow + k1, bRow + k1, &As[p ^ 1][lrow][lseg], &Bs[p ^ 1][lrow][lseg]);
        wait_async4();
      } else {
        wait_async0();
      }
      __syncthreads();

      v16bf afrag[2], bfrag[2];
#pragma unroll
      for (int mi = 0; mi < 2; ++mi) {
        const __bf16* ap = &As[p][waveM + mi * 16 + l16][0];
        afrag[mi] = make16(*(const v8bf*)(ap + 8 * hf),
                           *(const v8bf*)(ap + 16 + 8 * hf));
      }
#pragma unroll
      for (int ni = 0; ni < 2; ++ni) {
        const __bf16* bp = &Bs[p][waveN + ni * 16 + l16][0];
        bfrag[ni] = make16(*(const v8bf*)(bp + 8 * hf),
                           *(const v8bf*)(bp + 16 + 8 * hf));
      }
#pragma unroll
      for (int mi = 0; mi < 2; ++mi)
#pragma unroll
        for (int ni = 0; ni < 2; ++ni)
          acc[mi][ni] = __builtin_amdgcn_wmma_f32_16x16x32_bf16(
              false, afrag[mi], false, bfrag[ni], (short)0, acc[mi][ni],
              false, false);
      __syncthreads();
    }

    // running min update
#pragma unroll
    for (int mi = 0; mi < 2; ++mi)
#pragma unroll
      for (int ni = 0; ni < 2; ++ni) {
        const int col = n0 + waveN + ni * 16 + l16;
        const float cn = cbn[col];
#pragma unroll
        for (int v = 0; v < 8; ++v) {
          const float d = cn - 2.0f * acc[mi][ni][v];
          if (d < best[mi][v] || (d == best[mi][v] && col < bidx[mi][v])) {
            best[mi][v] = d; bidx[mi][v] = col;
          }
        }
      }
  }

  // cross-lane argmin over the 16 lanes (columns) sharing each row
#pragma unroll
  for (int mi = 0; mi < 2; ++mi)
#pragma unroll
    for (int v = 0; v < 8; ++v) {
      float bv = best[mi][v];
      int   bi = bidx[mi][v];
#pragma unroll
      for (int m = 1; m <= 8; m <<= 1) {
        float ov = __shfl_xor(bv, m, 32);
        int   oi = __shfl_xor(bi, m, 32);
        if (ov < bv || (ov == bv && oi < bi)) { bv = ov; bi = oi; }
      }
      if (l16 == 0)
        idxOut[bm + waveM + mi * 16 + hf * 8 + v] = bi;
    }
}

// ---------------------------------------------------------------------------
// Gather q = cb[idx] (f32 for loss, bf16 for decoder); accumulate
// BETA * mean((z - q)^2) into lossOut via atomic f32 add.
// ---------------------------------------------------------------------------
__global__ __launch_bounds__(256)
void gather_loss_kernel(const float* __restrict__ Z, const int* __restrict__ idx,
                        const float* __restrict__ cbf, __bf16* __restrict__ Qb,
                        float* __restrict__ lossOut, int rows, float scale)
{
  __shared__ float red[256];
  const size_t total = (size_t)rows * 256;
  float s = 0.0f;
  for (size_t e = (size_t)blockIdx.x * 256 + threadIdx.x; e < total;
       e += (size_t)gridDim.x * 256) {
    const int row = (int)(e >> 8);
    const int d   = (int)(e & 255);
    const int k   = idx[row];
    const float c = cbf[(size_t)k * 256 + d];
    Qb[e] = (__bf16)c;
    const float diff = Z[e] - c;
    s += diff * diff;
  }
  red[threadIdx.x] = s;
  __syncthreads();
  for (int off = 128; off > 0; off >>= 1) {
    if ((int)threadIdx.x < off) red[threadIdx.x] += red[threadIdx.x + off];
    __syncthreads();
  }
  if (threadIdx.x == 0) unsafeAtomicAdd(lossOut, red[0] * scale);
}

// ---------------------------------------------------------------------------
// Codebook prep: bf16 copy (padded rows = 0) + per-row squared norms
// (padded norms = +3e38 so padding never wins the argmin).
// ---------------------------------------------------------------------------
__global__ __launch_bounds__(64)
void build_cb_kernel(const float* __restrict__ cb, __bf16* __restrict__ cbb,
                     float* __restrict__ cbn, int K)
{
  __shared__ float red[64];
  const int k = blockIdx.x;
  float s = 0.0f;
  if (k < K) {
    for (int d = threadIdx.x; d < 256; d += 64) {
      const float v = cb[(size_t)k * 256 + d];
      cbb[(size_t)k * 256 + d] = (__bf16)v;
      s += v * v;
    }
  } else {
    for (int d = threadIdx.x; d < 256; d += 64)
      cbb[(size_t)k * 256 + d] = (__bf16)0.0f;
  }
  red[threadIdx.x] = s;
  __syncthreads();
  for (int off = 32; off > 0; off >>= 1) {
    if ((int)threadIdx.x < off) red[threadIdx.x] += red[threadIdx.x + off];
    __syncthreads();
  }
  if (threadIdx.x == 0) cbn[k] = (k < K) ? red[0] : 3.0e38f;
}

__global__ void cvt_bf16_kernel(const float* __restrict__ in,
                                __bf16* __restrict__ out, int n)
{
  for (int i = blockIdx.x * blockDim.x + threadIdx.x; i < n;
       i += gridDim.x * blockDim.x)
    out[i] = (__bf16)in[i];
}

// out (CxR) = transpose of in (RxC), in bf16: out[c*R + r] = in[r*C + c]
__global__ void cvt_t_bf16_kernel(const float* __restrict__ in,
                                  __bf16* __restrict__ out, int R, int C)
{
  const int n = R * C;
  for (int o = blockIdx.x * blockDim.x + threadIdx.x; o < n;
       o += gridDim.x * blockDim.x) {
    const int c = o / R, r = o % R;
    out[o] = (__bf16)in[(size_t)r * C + c];
  }
}

__global__ void zero_f32_kernel(float* p, int n)
{
  const int i = blockIdx.x * blockDim.x + threadIdx.x;
  if (i < n) p[i] = 0.0f;
}

// ---------------------------------------------------------------------------
// Orchestration
// ---------------------------------------------------------------------------
extern "C" void kernel_launch(void* const* d_in, const int* in_sizes, int n_in,
                              void* d_out, int out_size, void* d_ws, size_t ws_size,
                              hipStream_t stream)
{
  (void)in_sizes; (void)n_in; (void)out_size; (void)ws_size;
  const int B = 16384, D = 256;
  const int Ns[3]    = {3, 4, 3};
  const int Ksz[3]   = {500, 1000, 1000};
  const int Kpad[3]  = {512, 1024, 1024};
  const int xIdx[3]  = {0, 1, 2};
  const int pBase[3] = {3, 12, 21};   // t_W1.. / g_W1.. / e_W1..

  float* out     = (float*)d_out;
  float* lossPtr = out + (size_t)B * 3 * D;
  int*   idxBase = (int*)(lossPtr + 1);
  const int idxOff[3] = {0, B * 3, B * 3 + B * 4};

  // workspace carving (256B aligned)
  char* ws = (char*)d_ws;
  auto carve = [&](size_t bytes) -> char* {
    char* p = ws;
    ws += (bytes + 255) & ~(size_t)255;
    return p;
  };
  __bf16* xb   = (__bf16*)carve((size_t)B * D * 2);
  __bf16* hb   = (__bf16*)carve((size_t)B * D * 2);
  __bf16* dhb  = (__bf16*)carve((size_t)B * D * 2);
  __bf16* w1b  = (__bf16*)carve((size_t)D * D * 2);
  __bf16* w2b  = (__bf16*)carve((size_t)D * 1024 * 2);
  __bf16* dw1b = (__bf16*)carve((size_t)1024 * D * 2);
  __bf16* dw2b = (__bf16*)carve((size_t)D * D * 2);
  __bf16* cbb  = (__bf16*)carve((size_t)1024 * D * 2);
  float*  cbn  = (float*)carve((size_t)1024 * 4);
  float*  zf   = (float*)carve((size_t)B * 1024 * 4);
  __bf16* zb   = (__bf16*)carve((size_t)B * 1024 * 2);
  __bf16* qb   = (__bf16*)carve((size_t)B * 1024 * 2);

  zero_f32_kernel<<<1, 32, 0, stream>>>(lossPtr, 1);

  for (int s = 0; s < 3; ++s) {
    const int N = Ns[s], ND = N * D, K = Ksz[s], Kp = Kpad[s];
    const float* x   = (const float*)d_in[xIdx[s]];
    const float* W1  = (const float*)d_in[pBase[s] + 0];
    const float* b1  = (const float*)d_in[pBase[s] + 1];
    const float* W2  = (const float*)d_in[pBase[s] + 2];
    const float* b2  = (const float*)d_in[pBase[s] + 3];
    const float* cb  = (const float*)d_in[pBase[s] + 4];
    const float* dW1 = (const float*)d_in[pBase[s] + 5];
    const float* db1 = (const float*)d_in[pBase[s] + 6];
    const float* dW2 = (const float*)d_in[pBase[s] + 7];
    const float* db2 = (const float*)d_in[pBase[s] + 8];
    int* idxP = idxBase + idxOff[s];

    cvt_bf16_kernel<<<(B * D + 255) / 256, 256, 0, stream>>>(x, xb, B * D);
    cvt_t_bf16_kernel<<<(D * D + 255) / 256, 256, 0, stream>>>(W1, w1b, D, D);
    cvt_t_bf16_kernel<<<(D * ND + 255) / 256, 256, 0, stream>>>(W2, w2b, D, ND);
    cvt_t_bf16_kernel<<<(ND * D + 255) / 256, 256, 0, stream>>>(dW1, dw1b, ND, D);
    cvt_t_bf16_kernel<<<(D * D + 255) / 256, 256, 0, stream>>>(dW2, dw2b, D, D);
    build_cb_kernel<<<Kp, 64, 0, stream>>>(cb, cbb, cbn, K);

    // encoder: h = relu(x@W1 + b1)  (bf16 out only)
    gemm_bf16_wmma<<<dim3(D / BN, B / BM), 128, 0, stream>>>(
        xb, D, w1b, D, b1, nullptr, 0, hb, D, B, D, D, 1);
    // z = h@W2 + b2  (f32 for loss, bf16 for VQ)
    gemm_bf16_wmma<<<dim3(ND / BN, B / BM), 128, 0, stream>>>(
        hb, D, w2b, D, b2, zf, ND, zb, ND, B, ND, D, 0);
    // VQ argmin (fused distance GEMM + running argmin)
    vq_argmin_wmma<<<(B * N) / BM, 128, 0, stream>>>(zb, cbb, cbn, idxP, Kp);
    // gather q, accumulate BETA*mean((z-q)^2)
    gather_loss_kernel<<<1024, 256, 0, stream>>>(
        zf, idxP, cb, qb, lossPtr, B * N, 0.25f / (float)((size_t)B * N * D));
    // decoder: dh = relu(q@dW1 + db1)
    gemm_bf16_wmma<<<dim3(D / BN, B / BM), 128, 0, stream>>>(
        qb, ND, dw1b, ND, db1, nullptr, 0, dhb, D, B, D, ND, 1);
    // out[:, s, :] = dh@dW2 + db2  (f32 directly into stacked output)
    gemm_bf16_wmma<<<dim3(D / BN, B / BM), 128, 0, stream>>>(
        dhb, D, dw2b, D, db2, out + (size_t)s * D, 3 * D, nullptr, 0,
        B, D, D, 0);
  }
}
